// QRNNLayer_71219147702737
// MI455X (gfx1250) — compile-verified, matrix operand order checked by reading, not run on hardware
//
#include <hip/hip_runtime.h>
#include <math.h>

// ---------------------------------------------------------------------------
// QRNN (f/o-pool) for MI455X / gfx1250.
//   g = conv1d_causal(x, W, K=2) + b   -> bf16 WMMA GEMM, M=32768 N=1536 K=1024
//   z,f,o = tanh/sig/sig(split(g))     -> fused into GEMM epilogue
//   c_t = f*c + (1-f)*z ; h_t = c*o    -> fp32 sequential scan, 8192 lanes
// Round 2: 64x64 wave tile (16 WMMA : 16 b128-loads per K-step) and pure
// 32-bit per-lane offsets so loads lower to saddr+voffset form.
// ---------------------------------------------------------------------------

typedef __attribute__((ext_vector_type(16))) __bf16 v16bf;
typedef __attribute__((ext_vector_type(8)))  __bf16 v8bf;
typedef __attribute__((ext_vector_type(8)))  float  v8f;

#define T_SEQ 2048
#define B_SZ  16
#define CIN   512
#define H_SZ  512
#define N3    1536            // 3*H
#define KD    1024            // 2*CIN
#define M_ROWS (T_SEQ * B_SZ) // 32768

// workspace layout (bytes)
#define XBF_ELEMS ((T_SEQ + 1) * B_SZ * CIN)   // one leading zero time-block for x[-1]
#define XBF_BYTES ((size_t)XBF_ELEMS * 2)
#define WBF_ELEMS (N3 * KD)
#define WBF_BYTES ((size_t)WBF_ELEMS * 2)
#define G_OFF     (XBF_BYTES + WBF_BYTES)

union Frag { v16bf v; v8bf h[2]; };

// ---------------- x -> bf16, with one leading B*CIN zero block --------------
__global__ __launch_bounds__(256)
void cvt_x_kernel(const float* __restrict__ x, __bf16* __restrict__ xb) {
  int i = blockIdx.x * blockDim.x + threadIdx.x;
  if (i >= XBF_ELEMS) return;
  const int pad = B_SZ * CIN; // 8192
  xb[i] = (i < pad) ? (__bf16)0.0f : (__bf16)x[i - pad];
}

// ------- W (3H, Cin, K=2) -> Wbf (3H, 1024) bf16: Wbf[oc,k'] ----------------
// k' < 512  -> W[oc, k', 0]   (tap on x[t-1])
// k' >= 512 -> W[oc, k'-512, 1] (tap on x[t])
__global__ __launch_bounds__(256)
void cvt_w_kernel(const float* __restrict__ W, __bf16* __restrict__ wb) {
  int i = blockIdx.x * blockDim.x + threadIdx.x;
  if (i >= WBF_ELEMS) return;
  int oc = i >> 10;
  int kp = i & 1023;
  int src = (kp < CIN) ? (oc * KD + kp * 2) : (oc * KD + (kp - CIN) * 2 + 1);
  wb[i] = (__bf16)W[src];
}

// ---------------- GEMM + bias + activation (WMMA bf16) ----------------------
// Block 256 thr = 8 wave32 arranged 2(M) x 4(N); wave tile 64x64 (4x4 frags)
// -> block tile 128 x 256. grid = (M/128, N/256) = (256, 6).
__global__ __launch_bounds__(256)
void qrnn_gemm_kernel(const __bf16* __restrict__ Xbf,
                      const __bf16* __restrict__ Wbf,
                      const float*  __restrict__ bias,
                      float*        __restrict__ G) {
  const int lane     = threadIdx.x & 31;
  const int wave     = threadIdx.x >> 5;
  const int waveM    = wave >> 2;        // 0..1
  const int waveN    = wave & 3;         // 0..3
  const int mBase    = blockIdx.x * 128 + waveM * 64;
  const int nBase    = blockIdx.y * 256 + waveN * 64;
  const int laneRow  = lane & 15;
  const int laneHalf = lane >> 4;        // selects K-chunk pair per ISA layout

  const v8f vzero = {0.f,0.f,0.f,0.f,0.f,0.f,0.f,0.f};
  v8f acc[4][4];
#pragma unroll
  for (int i = 0; i < 4; ++i)
#pragma unroll
    for (int j = 0; j < 4; ++j) acc[i][j] = vzero;

  // 32-bit per-lane element offsets (all buffers < 2GB) -> saddr+voffset loads
  int aOff[4], bOff[4];
#pragma unroll
  for (int i = 0; i < 4; ++i)
    aOff[i] = (mBase + i * 16 + laneRow) * CIN + laneHalf * 8;
#pragma unroll
  for (int j = 0; j < 4; ++j)
    bOff[j] = (nBase + j * 16 + laneRow) * KD + laneHalf * 8;

  for (int kk = 0; kk < KD; kk += 32) {
    // A k' -> memory: k'<512 in block t (x[t-1]); k'>=512 in block t+1 (+8192)
    const int ao = (kk >= CIN) ? (B_SZ * CIN + (kk - CIN)) : kk;

    Frag aF[4], bF[4];
#pragma unroll
    for (int i = 0; i < 4; ++i) {
      aF[i].h[0] = *(const v8bf*)(Xbf + aOff[i] + ao);
      aF[i].h[1] = *(const v8bf*)(Xbf + aOff[i] + ao + 16);
    }
#pragma unroll
    for (int j = 0; j < 4; ++j) {
      bF[j].h[0] = *(const v8bf*)(Wbf + bOff[j] + kk);
      bF[j].h[1] = *(const v8bf*)(Wbf + bOff[j] + kk + 16);
    }

#pragma unroll
    for (int i = 0; i < 4; ++i)
#pragma unroll
      for (int j = 0; j < 4; ++j)
        acc[i][j] = __builtin_amdgcn_wmma_f32_16x16x32_bf16(
            false, aF[i].v, false, bF[j].v, (short)0, acc[i][j], false, false);
  }

  // Epilogue: bias + activation, store fp32 gates. C/D layout: VGPR r holds
  // M = r (+8 for lanes>=16), N = lane%16.
  const int col0 = lane & 15;
#pragma unroll
  for (int j = 0; j < 4; ++j) {
    const int col = nBase + j * 16 + col0;
    const float bc = bias[col];
    const bool isZ = (nBase + j * 16) < H_SZ;   // region 0 -> tanh, else sigmoid
#pragma unroll
    for (int i = 0; i < 4; ++i) {
      const int mRow = mBase + i * 16 + laneHalf * 8;
#pragma unroll
      for (int r = 0; r < 8; ++r) {
        float v = acc[i][j][r] + bc;
        v = isZ ? tanhf(v) : (1.0f / (1.0f + __expf(-v)));
        G[(mRow + r) * N3 + col] = v;
      }
    }
  }
}

// ---------------- sequential scan: 8192 independent (b,h) lanes -------------
__global__ __launch_bounds__(256)
void qrnn_scan_kernel(const float* __restrict__ G, float* __restrict__ out) {
  const int u = blockIdx.x * blockDim.x + threadIdx.x;   // 0..8191
  const int b = u >> 9;
  const int h = u & 511;
  const float* gz = G + b * N3 + h;                      // z; f at +512; o at +1024
  float* hp = out + b * H_SZ + h;
  float c = 0.0f, hv = 0.0f;
  for (int t = 0; t < T_SEQ; ++t) {
    const int off = t * (B_SZ * N3);
    const float z = gz[off];
    const float f = gz[off + H_SZ];
    const float o = gz[off + 2 * H_SZ];
    c  = f * c + (1.0f - f) * z;
    hv = c * o;
    hp[t * (B_SZ * H_SZ)] = hv;
  }
  const size_t hsEnd = (size_t)T_SEQ * B_SZ * H_SZ;      // 16777216
  out[hsEnd + u]               = hv;                     // h_last (1,B,H)
  out[hsEnd + B_SZ * H_SZ + u] = c;                      // c_last (1,B,H)
}

// ---------------------------------------------------------------------------
extern "C" void kernel_launch(void* const* d_in, const int* in_sizes, int n_in,
                              void* d_out, int out_size, void* d_ws, size_t ws_size,
                              hipStream_t stream) {
  const float* x    = (const float*)d_in[0];
  const float* W    = (const float*)d_in[1];
  const float* bias = (const float*)d_in[2];
  float* out = (float*)d_out;

  char* ws = (char*)d_ws;
  __bf16* Xbf = (__bf16*)ws;
  __bf16* Wbf = (__bf16*)(ws + XBF_BYTES);
  float*  G   = (float*)(ws + G_OFF);

  cvt_x_kernel<<<(XBF_ELEMS + 255) / 256, 256, 0, stream>>>(x, Xbf);
  cvt_w_kernel<<<(WBF_ELEMS + 255) / 256, 256, 0, stream>>>(W, Wbf);
  qrnn_gemm_kernel<<<dim3(M_ROWS / 128, N3 / 256), 256, 0, stream>>>(Xbf, Wbf, bias, G);
  qrnn_scan_kernel<<<(B_SZ * H_SZ) / 256, 256, 0, stream>>>(G, out);
}
